// Qwen3MoeMLP_47691316855583
// MI455X (gfx1250) — matrix operand order repaired and checked by reading, not compiled, not run on hardware
//
#include <hip/hip_runtime.h>
#include <stdint.h>

#define N_TOK 16384
#define HID   2048
#define INTER 768
#define TOPK  8

typedef __attribute__((ext_vector_type(16))) __bf16 v16bf;
typedef __attribute__((ext_vector_type(8)))  float  v8f;
typedef __attribute__((ext_vector_type(8)))  unsigned int v8u;

// round-to-nearest-even fp32 -> bf16 (bit pattern)
__device__ __forceinline__ unsigned short f2bf(float f) {
  unsigned int u = __builtin_bit_cast(unsigned int, f);
  unsigned int r = u + 0x7FFFu + ((u >> 16) & 1u);
  return (unsigned short)(r >> 16);
}
__device__ __forceinline__ unsigned int pack2bf(float a, float b) {
  return (unsigned int)f2bf(a) | ((unsigned int)f2bf(b) << 16);
}

// Build a 16-element bf16 fragment from two 16-byte LDS chunks.
__device__ __forceinline__ v16bf make_frag(const unsigned short* p0,
                                           const unsigned short* p1) {
  const uint4 a = *(const uint4*)p0;
  const uint4 b = *(const uint4*)p1;
  v8u t;
  t[0] = a.x; t[1] = a.y; t[2] = a.z; t[3] = a.w;
  t[4] = b.x; t[5] = b.y; t[6] = b.z; t[7] = b.w;
  return __builtin_bit_cast(v16bf, t);
}
// A fragment (16x32 bf16): lane<16 -> K 0..15; lane>=16 -> K 16..31 (contiguous 32B)
__device__ __forceinline__ v16bf frag_a(const unsigned short* rowbase) {
  return make_frag(rowbase, rowbase + 8);
}
// B fragment (32x16 bf16): lane<16 -> K 0..15; lane>=16 -> K 16..31 (contiguous 32B)
__device__ __forceinline__ v16bf frag_b(const unsigned short* rowbase) {
  return make_frag(rowbase, rowbase + 8);
}

// ---- CDNA5 async global->LDS copy (ASYNCcnt-tracked), 16 bytes per lane ----
__device__ __forceinline__ void async_b128(const unsigned short* g,
                                           const unsigned short* l) {
  unsigned lds_off = (unsigned)(uintptr_t)l;          // low 32 bits = LDS offset
  unsigned long long ga = (unsigned long long)(uintptr_t)g;
  asm volatile("global_load_async_to_lds_b128 %0, %1, off"
               :: "v"(lds_off), "v"(ga) : "memory");
}

// ---------------- kernel: fp32 -> bf16 bulk convert -----------------------
__global__ __launch_bounds__(256)
void cvt_bf16(const float* __restrict__ src, unsigned short* __restrict__ dst,
              int n4) {
  int i = blockIdx.x * 256 + threadIdx.x;
  if (i < n4) {
    const float4 v = *(const float4*)(src + (size_t)i * 4);
    uint2 p;
    p.x = pack2bf(v.x, v.y);
    p.y = pack2bf(v.z, v.w);
    *(uint2*)(dst + (size_t)i * 4) = p;
  }
}

// ---------------- kernel: routing weights ---------------------------------
__global__ void moe_routing_w(const int* __restrict__ mask,
                              const float* __restrict__ rw,
                              float* __restrict__ wout) {
  int k = blockIdx.x * blockDim.x + threadIdx.x;
  if (k < N_TOK) {
    float s = 0.f;
#pragma unroll
    for (int j = 0; j < TOPK; ++j)
      s += (float)mask[j * N_TOK + k] * rw[k * TOPK + j];
    wout[k] = s;
  }
}

// ---------------- kernel 1: gate+up GEMM, SwiGLU, bf16 act ----------------
// WG tile: 128 tokens x 64 intermediate cols; K-step 64; double-buffered LDS.
template <bool XBF16>
__global__ __launch_bounds__(256)
void moe_gateup(const float* __restrict__ Xf,
                const unsigned short* __restrict__ Xb,
                const unsigned short* __restrict__ gwb,
                const unsigned short* __restrict__ uwb,
                unsigned short* __restrict__ act) {
  __shared__ __align__(16) unsigned short As[2][128 * 72];
  __shared__ __align__(16) unsigned short Bg[2][64 * 72];
  __shared__ __align__(16) unsigned short Bu[2][64 * 72];

  const int tid  = threadIdx.x;
  const int lane = tid & 31;
  const int wid  = tid >> 5;
  const int lm   = lane & 15;
  const int kh   = lane >> 4;
  const int m0   = blockIdx.y * 128;
  const int n0   = blockIdx.x * 64;
  const int wm   = (wid & 3) * 32;
  const int wn   = (wid >> 2) * 32;

  v8f accg[2][2], accu[2][2];
#pragma unroll
  for (int i = 0; i < 2; ++i)
#pragma unroll
    for (int j = 0; j < 2; ++j)
#pragma unroll
      for (int e = 0; e < 8; ++e) { accg[i][j][e] = 0.f; accu[i][j][e] = 0.f; }

  auto stage = [&](int buf, int k0) {
    { // B tiles: thread -> (row = tid>>2, 32B quarter = tid&3), 2 asyncs each
      int r = tid >> 2, q = (tid & 3) * 16;
      const unsigned short* gg = gwb + (size_t)(n0 + r) * HID + k0 + q;
      const unsigned short* gu = uwb + (size_t)(n0 + r) * HID + k0 + q;
      unsigned short* lg = &Bg[buf][r * 72 + q];
      unsigned short* lu = &Bu[buf][r * 72 + q];
      async_b128(gg, lg); async_b128(gg + 8, lg + 8);
      async_b128(gu, lu); async_b128(gu + 8, lu + 8);
    }
    if constexpr (XBF16) { // A tile: thread -> (row = tid>>1, 64B half), 4 asyncs
      int r = tid >> 1, h = (tid & 1) * 32;
      const unsigned short* g = Xb + (size_t)(m0 + r) * HID + k0 + h;
      unsigned short* l = &As[buf][r * 72 + h];
#pragma unroll
      for (int j = 0; j < 4; ++j) async_b128(g + j * 8, l + j * 8);
    } else { // fallback: convert fp32 X on the fly
#pragma unroll
      for (int i = tid; i < 4096; i += 256) {
        int r = i >> 5, c = (i & 31) << 1;
        const float2 v = *(const float2*)(Xf + (size_t)(m0 + r) * HID + k0 + c);
        *(unsigned int*)&As[buf][r * 72 + c] = pack2bf(v.x, v.y);
      }
    }
  };

  constexpr int NK = HID / 64;
  stage(0, 0);
  for (int i = 0; i < NK; ++i) {
    const int cur = i & 1;
    if (i + 1 < NK) {
      stage(cur ^ 1, (i + 1) * 64);
      if constexpr (XBF16) asm volatile("s_wait_asynccnt 0x8" ::: "memory");
      else                 asm volatile("s_wait_asynccnt 0x4" ::: "memory");
    } else {
      asm volatile("s_wait_asynccnt 0x0" ::: "memory");
    }
    __syncthreads();

#pragma unroll
    for (int kk = 0; kk < 64; kk += 32) {
      v16bf afr[2], bgf[2], buf2[2];
#pragma unroll
      for (int tm = 0; tm < 2; ++tm)
        afr[tm] = frag_a(&As[cur][(wm + tm * 16 + lm) * 72 + kk + kh * 16]);
#pragma unroll
      for (int tn = 0; tn < 2; ++tn) {
        bgf[tn]  = frag_b(&Bg[cur][(wn + tn * 16 + lm) * 72 + kk + kh * 16]);
        buf2[tn] = frag_b(&Bu[cur][(wn + tn * 16 + lm) * 72 + kk + kh * 16]);
      }
#pragma unroll
      for (int tm = 0; tm < 2; ++tm)
#pragma unroll
        for (int tn = 0; tn < 2; ++tn) {
          accg[tm][tn] = __builtin_amdgcn_wmma_f32_16x16x32_bf16(
              false, afr[tm], false, bgf[tn], (short)0, accg[tm][tn], false, false);
          accu[tm][tn] = __builtin_amdgcn_wmma_f32_16x16x32_bf16(
              false, afr[tm], false, buf2[tn], (short)0, accu[tm][tn], false, false);
        }
    }
    __syncthreads();
  }

  // SwiGLU epilogue -> bf16 act
#pragma unroll
  for (int tm = 0; tm < 2; ++tm)
#pragma unroll
    for (int tn = 0; tn < 2; ++tn) {
      int col = n0 + wn + tn * 16 + lm;
#pragma unroll
      for (int r = 0; r < 8; ++r) {
        int row = m0 + wm + tm * 16 + kh * 8 + r;
        float g = accg[tm][tn][r];
        float u = accu[tm][tn][r];
        float s = g / (1.0f + __expf(-g));
        act[(size_t)row * INTER + col] = f2bf(s * u);
      }
    }
}

// ---------------- kernel 2: down GEMM + scale + residual ------------------
// WG tile: 128 tokens x 64 hidden cols; K-step 64; double-buffered LDS.
__global__ __launch_bounds__(256)
void moe_down(const unsigned short* __restrict__ act,
              const unsigned short* __restrict__ dwb,
              const float* __restrict__ wtok,
              const float* __restrict__ fin,
              float* __restrict__ out) {
  __shared__ __align__(16) unsigned short As[2][128 * 72];
  __shared__ __align__(16) unsigned short Bs[2][64 * 72];

  const int tid  = threadIdx.x;
  const int lane = tid & 31;
  const int wid  = tid >> 5;
  const int lm   = lane & 15;
  const int kh   = lane >> 4;
  const int m0   = blockIdx.y * 128;
  const int n0   = blockIdx.x * 64;
  const int wm   = (wid & 3) * 32;
  const int wn   = (wid >> 2) * 32;

  v8f acc[2][2];
#pragma unroll
  for (int i = 0; i < 2; ++i)
#pragma unroll
    for (int j = 0; j < 2; ++j)
#pragma unroll
      for (int e = 0; e < 8; ++e) acc[i][j][e] = 0.f;

  auto stage = [&](int buf, int k0) {
    { // down_w tile (bf16): 2 asyncs
      int r = tid >> 2, q = (tid & 3) * 16;
      const unsigned short* g = dwb + (size_t)(n0 + r) * INTER + k0 + q;
      unsigned short* l = &Bs[buf][r * 72 + q];
      async_b128(g, l); async_b128(g + 8, l + 8);
    }
    { // act tile (bf16): 4 asyncs
      int r = tid >> 1, h = (tid & 1) * 32;
      const unsigned short* g = act + (size_t)(m0 + r) * INTER + k0 + h;
      unsigned short* l = &As[buf][r * 72 + h];
#pragma unroll
      for (int j = 0; j < 4; ++j) async_b128(g + j * 8, l + j * 8);
    }
  };

  constexpr int NK = INTER / 64;
  stage(0, 0);
  for (int i = 0; i < NK; ++i) {
    const int cur = i & 1;
    if (i + 1 < NK) {
      stage(cur ^ 1, (i + 1) * 64);
      asm volatile("s_wait_asynccnt 0x6" ::: "memory");
    } else {
      asm volatile("s_wait_asynccnt 0x0" ::: "memory");
    }
    __syncthreads();

#pragma unroll
    for (int kk = 0; kk < 64; kk += 32) {
      v16bf afr[2], bfr[2];
#pragma unroll
      for (int tm = 0; tm < 2; ++tm)
        afr[tm] = frag_a(&As[cur][(wm + tm * 16 + lm) * 72 + kk + kh * 16]);
#pragma unroll
      for (int tn = 0; tn < 2; ++tn)
        bfr[tn] = frag_b(&Bs[cur][(wn + tn * 16 + lm) * 72 + kk + kh * 16]);
#pragma unroll
      for (int tm = 0; tm < 2; ++tm)
#pragma unroll
        for (int tn = 0; tn < 2; ++tn)
          acc[tm][tn] = __builtin_amdgcn_wmma_f32_16x16x32_bf16(
              false, afr[tm], false, bfr[tn], (short)0, acc[tm][tn], false, false);
    }
    __syncthreads();
  }

  // epilogue: out = final + y * w[token]
#pragma unroll
  for (int tm = 0; tm < 2; ++tm)
#pragma unroll
    for (int tn = 0; tn < 2; ++tn) {
      int col = n0 + wn + tn * 16 + lm;
#pragma unroll
      for (int r = 0; r < 8; ++r) {
        int row = m0 + wm + tm * 16 + kh * 8 + r;
        size_t idx = (size_t)row * HID + col;
        out[idx] = fin[idx] + acc[tm][tn][r] * wtok[row];
      }
    }
}

extern "C" void kernel_launch(void* const* d_in, const int* in_sizes, int n_in,
                              void* d_out, int out_size, void* d_ws, size_t ws_size,
                              hipStream_t stream) {
  const float* X    = (const float*)d_in[0];  // hidden_states [N,H]
  const int*   mask = (const int*)  d_in[1];  // expert_mask   [TOPK,N]
  const float* rw   = (const float*)d_in[2];  // routing_weights [N,TOPK]
  const float* fin  = (const float*)d_in[3];  // final_hidden_states [N,H]
  const float* gw   = (const float*)d_in[4];  // gate_w [I,H]
  const float* uw   = (const float*)d_in[5];  // up_w   [I,H]
  const float* dw   = (const float*)d_in[6];  // down_w [H,I]
  float* out = (float*)d_out;

  // workspace layout (256-B aligned slices)
  size_t off = 0;
  auto take = [&](size_t bytes) {
    size_t o = off;
    off = (off + bytes + 255) & ~(size_t)255;
    return o;
  };
  char* ws = (char*)d_ws;
  const size_t IH = (size_t)INTER * HID;      // 1.57M elems per weight
  float*          wtok = (float*)         (ws + take((size_t)N_TOK * 4));
  unsigned short* gwb  = (unsigned short*)(ws + take(IH * 2));
  unsigned short* uwb  = (unsigned short*)(ws + take(IH * 2));
  unsigned short* dwb  = (unsigned short*)(ws + take(IH * 2));
  unsigned short* act  = (unsigned short*)(ws + take((size_t)N_TOK * INTER * 2));
  const size_t xb_bytes = (size_t)N_TOK * HID * 2;
  const bool xbf16 = (off + xb_bytes) <= ws_size;
  unsigned short* Xb = (unsigned short*)(ws + take(xb_bytes)); // valid iff xbf16

  moe_routing_w<<<dim3((N_TOK + 255) / 256), dim3(256), 0, stream>>>(mask, rw, wtok);

  const int wq = (int)(IH / 4);               // quads per weight matrix
  cvt_bf16<<<dim3((wq + 255) / 256), dim3(256), 0, stream>>>(gw, gwb, wq);
  cvt_bf16<<<dim3((wq + 255) / 256), dim3(256), 0, stream>>>(uw, uwb, wq);
  cvt_bf16<<<dim3((wq + 255) / 256), dim3(256), 0, stream>>>(dw, dwb, wq);
  if (xbf16) {
    const int xq = (int)((size_t)N_TOK * HID / 4);
    cvt_bf16<<<dim3((xq + 255) / 256), dim3(256), 0, stream>>>(X, Xb, xq);
    moe_gateup<true><<<dim3(INTER / 64, N_TOK / 128), dim3(256), 0, stream>>>(
        X, Xb, gwb, uwb, act);
  } else {
    moe_gateup<false><<<dim3(INTER / 64, N_TOK / 128), dim3(256), 0, stream>>>(
        X, (const unsigned short*)nullptr, gwb, uwb, act);
  }
  moe_down<<<dim3(HID / 64, N_TOK / 128), dim3(256), 0, stream>>>(
      act, dwb, wtok, fin, out);
}
